// KANLayer_52450140619301
// MI455X (gfx1250) — compile-verified, matrix operand order checked by reading, not run on hardware
//
#include <hip/hip_runtime.h>
#include <hip/hip_bf16.h>

typedef _Float16 v16h __attribute__((ext_vector_type(16)));
typedef _Float16 v8h  __attribute__((ext_vector_type(8)));
typedef _Float16 v4h  __attribute__((ext_vector_type(4)));
typedef float    v8f  __attribute__((ext_vector_type(8)));

#define BATCH 8192
#define INF   2048   // K
#define OUTF  2048   // N

#define BM 128
#define BN 128
#define BK 32
#define LDSS 56      // LDS row stride in halves: 112B, 16B aligned, conflict-free bank striding

#if defined(__AMDGCN__) && __has_builtin(__builtin_amdgcn_global_load_async_to_lds_b128) && \
    __has_builtin(__builtin_amdgcn_s_wait_asynccnt)
#define USE_ASYNC 1
#else
#define USE_ASYNC 0
#endif

#if USE_ASYNC
typedef int async_v4i __attribute__((vector_size(16)));
typedef __attribute__((address_space(1))) async_v4i* async_gptr;   // printed as "v4i __device__ *"
typedef __attribute__((address_space(3))) async_v4i* async_lptr;   // printed as "v4i __shared__ *"
#endif

// ---------------- prep kernels ----------------

__global__ void tanh_to_f16(const float* __restrict__ x, _Float16* __restrict__ T, int n4) {
    int idx = blockIdx.x * blockDim.x + threadIdx.x;
    if (idx >= n4) return;
    const float4 v = *(const float4*)(x + (size_t)idx * 4);
    v4h o;
    o[0] = (_Float16)tanhf(v.x);
    o[1] = (_Float16)tanhf(v.y);
    o[2] = (_Float16)tanhf(v.z);
    o[3] = (_Float16)tanhf(v.w);
    *(v4h*)(T + (size_t)idx * 4) = o;
}

// W[k][n] f32 -> Wt[n][k] f16 (tiled LDS transpose, done once)
__global__ void transpose_w_f16(const float* __restrict__ W, _Float16* __restrict__ Wt) {
    __shared__ _Float16 t[32][33];
    const int tx = threadIdx.x;            // 0..31
    const int ty = threadIdx.y;            // 0..7
    const int n0 = blockIdx.x * 32;
    const int k0 = blockIdx.y * 32;
    #pragma unroll
    for (int j = 0; j < 32; j += 8)
        t[ty + j][tx] = (_Float16)W[(size_t)(k0 + ty + j) * OUTF + n0 + tx];
    __syncthreads();
    #pragma unroll
    for (int j = 0; j < 32; j += 8)
        Wt[(size_t)(n0 + ty + j) * INF + k0 + tx] = t[tx][ty + j];
}

// partial column sums of B: grid (OUTF/256, 16), each block sums 128 rows
__global__ void colsum_partial(const float* __restrict__ Bm, float* __restrict__ part) {
    int j = blockIdx.x * blockDim.x + threadIdx.x;
    const float* p = Bm + (size_t)blockIdx.y * 128 * OUTF + j;
    float s = 0.f;
    #pragma unroll 8
    for (int i = 0; i < 128; ++i) s += p[(size_t)i * OUTF];
    part[(size_t)blockIdx.y * OUTF + j] = s;
}

__global__ void colsum_reduce(const float* __restrict__ part, float* __restrict__ bias) {
    int j = blockIdx.x * blockDim.x + threadIdx.x;
    float s = 0.f;
    #pragma unroll
    for (int c = 0; c < 16; ++c) s += part[(size_t)c * OUTF + j];
    bias[j] = s;
}

// ---------------- WMMA GEMM ----------------

// per-lane 16B global -> LDS copy (async DMA when available)
static __device__ __forceinline__ void cp_b128(const _Float16* g, _Float16* l) {
#if USE_ASYNC
    __builtin_amdgcn_global_load_async_to_lds_b128(
        (async_gptr)(void*)const_cast<_Float16*>(g),
        (async_lptr)(void*)l,
        0, 0);
#else
    *(v8h*)l = *(const v8h*)g;
#endif
}

// 16x32 f16 A/B fragment (v16h) from an LDS row (ISA layout):
// half-group h: elems 0..7 = K 8h..8h+7 ; elems 8..15 = K 16+8h..16+8h+7
static __device__ __forceinline__ v16h ld_frag(const _Float16* rowp, int h) {
    v8h lo = *(const v8h*)(rowp + 8 * h);
    v8h hi = *(const v8h*)(rowp + 16 + 8 * h);
    return __builtin_shufflevector(lo, hi, 0,1,2,3,4,5,6,7,8,9,10,11,12,13,14,15);
}

__global__ __launch_bounds__(256)
void kan_gemm(const _Float16* __restrict__ T, const _Float16* __restrict__ Wt,
              const float* __restrict__ bias, float* __restrict__ out) {
    __shared__ _Float16 As[2][BM * LDSS];   // tanh(x) tile, [m][k]
    __shared__ _Float16 Bs[2][BN * LDSS];   // Wt tile,      [n][k]

    const int tid = threadIdx.x;
    const int m0  = blockIdx.y * BM;
    const int n0  = blockIdx.x * BN;

    // staging: each thread moves 4x 16B per tile (2 A rows + 2 B rows)
    const int srow = tid >> 2;             // 0..63  (+64 for second row)
    const int skof = (tid & 3) * 8;        // 8 halves = 16B
    const _Float16* gA = T  + (size_t)(m0 + srow) * INF + skof;
    const _Float16* gB = Wt + (size_t)(n0 + srow) * INF + skof;
    const size_t rstep = (size_t)64 * INF;

    // wave tiling: 8 waves -> 2 (M) x 4 (N); wave tile 64x32
    const int lane = tid & 31;
    const int wid  = tid >> 5;
    const int r    = lane & 15;
    const int h    = lane >> 4;
    const int wm   = (wid & 1) * 64;
    const int wn   = (wid >> 1) * 32;

    const v8f vzero = {0.f,0.f,0.f,0.f,0.f,0.f,0.f,0.f};
    v8f acc[4][2];
    #pragma unroll
    for (int mi = 0; mi < 4; ++mi)
        #pragma unroll
        for (int ni = 0; ni < 2; ++ni) acc[mi][ni] = vzero;

    const int nTiles = INF / BK;           // 64

    // stage tile 0 into buffer 0
    {
        cp_b128(gA,         &As[0][srow * LDSS + skof]);
        cp_b128(gA + rstep, &As[0][(srow + 64) * LDSS + skof]);
        cp_b128(gB,         &Bs[0][srow * LDSS + skof]);
        cp_b128(gB + rstep, &Bs[0][(srow + 64) * LDSS + skof]);
    }

    for (int i = 0; i < nTiles; ++i) {
        const int cur = i & 1;
#if USE_ASYNC
        __builtin_amdgcn_s_wait_asynccnt(0);   // my tile-i DMAs done
#endif
        __syncthreads();                        // tile i visible; buffer cur^1 free

        if (i + 1 < nTiles) {                   // stage tile i+1 (overlaps compute)
            const int nxt = cur ^ 1;
            const int kk  = (i + 1) * BK;
            cp_b128(gA + kk,         &As[nxt][srow * LDSS + skof]);
            cp_b128(gA + kk + rstep, &As[nxt][(srow + 64) * LDSS + skof]);
            cp_b128(gB + kk,         &Bs[nxt][srow * LDSS + skof]);
            cp_b128(gB + kk + rstep, &Bs[nxt][(srow + 64) * LDSS + skof]);
            if (i + 2 < nTiles) {               // prefetch tile i+2 toward near caches
                __builtin_prefetch(gA + kk + BK, 0, 3);
                __builtin_prefetch(gA + kk + BK + rstep, 0, 3);
                __builtin_prefetch(gB + kk + BK, 0, 3);
                __builtin_prefetch(gB + kk + BK + rstep, 0, 3);
            }
        }

        // compute on buffer cur
        v16h bf[2];
        #pragma unroll
        for (int ni = 0; ni < 2; ++ni)
            bf[ni] = ld_frag(&Bs[cur][(wn + ni * 16 + r) * LDSS], h);

        #pragma unroll
        for (int mi = 0; mi < 4; ++mi) {
            v16h af = ld_frag(&As[cur][(wm + mi * 16 + r) * LDSS], h);
            #pragma unroll
            for (int ni = 0; ni < 2; ++ni) {
                acc[mi][ni] = __builtin_amdgcn_wmma_f32_16x16x32_f16(
                    false, af, false, bf[ni], (short)0, acc[mi][ni], false, false);
            }
        }
    }

    // epilogue: add bias column sums, store f32
    const float b0 = bias[n0 + wn +      r];
    const float b1 = bias[n0 + wn + 16 + r];
    #pragma unroll
    for (int mi = 0; mi < 4; ++mi) {
        #pragma unroll
        for (int ni = 0; ni < 2; ++ni) {
            const int rowbase = m0 + wm + mi * 16 + 8 * h;
            const int col     = n0 + wn + ni * 16 + r;
            const float bb    = ni ? b1 : b0;
            #pragma unroll
            for (int v = 0; v < 8; ++v)
                out[(size_t)(rowbase + v) * OUTF + col] = acc[mi][ni][v] + bb;
        }
    }
}

// ---------------- launcher ----------------

extern "C" void kernel_launch(void* const* d_in, const int* in_sizes, int n_in,
                              void* d_out, int out_size, void* d_ws, size_t ws_size,
                              hipStream_t stream) {
    const float* x  = (const float*)d_in[0];
    const float* W  = (const float*)d_in[1];
    const float* Bm = (const float*)d_in[2];
    float* out = (float*)d_out;

    char* ws = (char*)d_ws;
    _Float16* T    = (_Float16*)(ws);                       // 8192*2048*2 = 32 MiB
    _Float16* Wt   = (_Float16*)(ws + (size_t)33554432);    // 2048*2048*2 =  8 MiB
    float*    part = (float*)   (ws + (size_t)41943040);    // 16*2048*4  = 128 KiB
    float*    bias = (float*)   (ws + (size_t)42074112);    // 2048*4     =   8 KiB

    // tanh(x) -> f16
    {
        int n4 = (BATCH * INF) / 4;
        tanh_to_f16<<<n4 / 256, 256, 0, stream>>>(x, T, n4);
    }
    // W -> Wt (f16, transposed [n][k])
    {
        dim3 g(OUTF / 32, INF / 32);
        transpose_w_f16<<<g, dim3(32, 8), 0, stream>>>(W, Wt);
    }
    // deterministic column sums of B
    {
        dim3 g(OUTF / 256, 16);
        colsum_partial<<<g, 256, 0, stream>>>(Bm, part);
        colsum_reduce<<<OUTF / 256, 256, 0, stream>>>(part, bias);
    }
    // WMMA GEMM: out = T @ Wt^T + bias
    {
        dim3 g(OUTF / BN, BATCH / BM);
        kan_gemm<<<g, 256, 0, stream>>>(T, Wt, bias, out);
    }
}
